// CrossmodalMultiHeadAttention_12025908429125
// MI455X (gfx1250) — compile-verified, matrix operand order checked by reading, not run on hardware
//
#include <hip/hip_runtime.h>

// ---------- types ----------
typedef __bf16 v16bf __attribute__((ext_vector_type(16)));
typedef float  v8f   __attribute__((ext_vector_type(8)));
typedef unsigned short ushort8 __attribute__((ext_vector_type(8)));
typedef unsigned short ushort4v __attribute__((ext_vector_type(4)));
typedef unsigned short u16;

union Frag {
    v16bf v;
    ushort8 h[2];
};

__device__ __forceinline__ u16 f2bf(float f) {
    unsigned u = __builtin_bit_cast(unsigned, f);
    // round-to-nearest-even truncation to bf16
    return (u16)((u + 0x7FFFu + ((u >> 16) & 1u)) >> 16);
}

__device__ __forceinline__ v8f v8zero() {
    v8f z = {0.f, 0.f, 0.f, 0.f, 0.f, 0.f, 0.f, 0.f};
    return z;
}

__device__ __forceinline__ v8f wmma_bf16(const Frag& a, const Frag& b, v8f c) {
    // D = A(16x32 bf16) * B(32x16 bf16) + C(16x16 f32)
    return __builtin_amdgcn_wmma_f32_16x16x32_bf16(
        /*neg_a=*/false, a.v, /*neg_b=*/false, b.v,
        /*c_mod=*/(short)0, c, /*reuse_a=*/false, /*reuse_b=*/false);
}

// 16B async copy: global -> LDS, tracked with ASYNCcnt (CDNA5 path).
// ldsOff = raw LDS byte offset (low 32 bits of a generic pointer to __shared__),
// gaddr  = per-lane 64-bit global address.
__device__ __forceinline__ void async_g2l_b128(unsigned ldsOff, const void* gaddr) {
    asm volatile("global_load_async_to_lds_b128 %0, %1, off"
                 :
                 : "v"(ldsOff), "v"((unsigned long long)(uintptr_t)gaddr)
                 : "memory");
}
__device__ __forceinline__ void wait_async0() {
    asm volatile("s_wait_asynccnt 0x0" ::: "memory");
}

// =====================================================================
// Generic GEMM:  out[M,N] = A[M,K] @ W[K,N] + bias
// A is f32 (converted to bf16 while staging) or bf16 (async-copied to LDS).
// out is f32 (final projection) or bf16 (intermediate in workspace).
// Block: 256 threads (8 waves). Tile: 128(M) x 128(N), K-step 32.
// Wave grid 4(M) x 2(N): each wave owns a 32x64 sub-tile -> 8 WMMA / K-step.
// =====================================================================
template <bool A_IS_F32, bool OUT_F32>
__global__ __launch_bounds__(256)
void gemm_wmma_kernel(const void* __restrict__ Av,
                      const float* __restrict__ W,
                      const float* __restrict__ bias,
                      void* __restrict__ outv,
                      int M, int N, int K) {
    __shared__ u16 As[128 * 40];  // [row][k], stride 40 halves (80B, 16B aligned)
    __shared__ u16 Bs[128 * 40];  // transposed W tile: [n][k]

    const int tid   = threadIdx.x;
    const int wave  = tid >> 5;
    const int lane  = tid & 31;
    const int laneM = lane & 15;
    const int hi    = lane >> 4;     // 0: K 0..7/16..23   1: K 8..15/24..31
    const int wm    = wave & 3;      // wave M index (0..3)  -> rows wm*32..+31
    const int wn    = wave >> 2;     // wave N index (0..1)  -> cols wn*64..+63

    const int m0 = blockIdx.y * 128;
    const int n0 = blockIdx.x * 128;

    v8f acc[2][4];
    for (int s = 0; s < 2; ++s)
        for (int n = 0; n < 4; ++n) acc[s][n] = v8zero();

    for (int k0 = 0; k0 < K; k0 += 32) {
        // ---- stage A tile (128 x 32) ----
        if (A_IS_F32) {
            const float* A = (const float*)Av;
            for (int i = 0; i < 4; ++i) {
                int ch  = tid + i * 256;          // 1024 float4 chunks
                int row = ch >> 3;
                int off = (ch & 7) << 2;
                const float4 f = *(const float4*)(A + (size_t)(m0 + row) * K + k0 + off);
                ushort4v hv;
                hv.x = f2bf(f.x); hv.y = f2bf(f.y);
                hv.z = f2bf(f.z); hv.w = f2bf(f.w);
                *(ushort4v*)(&As[row * 40 + off]) = hv;
            }
        } else {
            // A already bf16: asynchronous copy straight into LDS (ASYNCcnt path)
            const u16* A = (const u16*)Av;
            for (int i = 0; i < 2; ++i) {
                int ch  = tid + i * 256;          // 512 8-half chunks
                int row = ch >> 2;
                int off = (ch & 3) << 3;
                async_g2l_b128((unsigned)(uintptr_t)&As[row * 40 + off],
                               A + (size_t)(m0 + row) * K + k0 + off);
            }
        }
        // ---- stage B tile (32 x 128), transposed into Bs[n][k] ----
        for (int i = 0; i < 16; ++i) {
            int lin = tid + i * 256;              // 4096 elements
            int nn  = lin & 127;
            int kk  = lin >> 7;
            Bs[nn * 40 + kk] = f2bf(W[(size_t)(k0 + kk) * N + n0 + nn]);
        }
        // prefetch next W tile toward L2 while this K-step computes
        if (k0 + 32 < K)
            __builtin_prefetch(&W[(size_t)(k0 + 32 + (tid >> 7) * 16) * N + n0 + (tid & 127)], 0, 1);

        if (!A_IS_F32) wait_async0();
        __syncthreads();

        // ---- compute: each wave does a 32(M) x 64(N) sub-tile ----
        Frag a[2];
        for (int s = 0; s < 2; ++s) {
            const u16* ap = &As[(wm * 32 + s * 16 + laneM) * 40 + hi * 8];
            a[s].h[0] = *(const ushort8*)(ap);
            a[s].h[1] = *(const ushort8*)(ap + 16);
        }
        for (int n = 0; n < 4; ++n) {
            Frag b;
            const u16* bp = &Bs[(wn * 64 + n * 16 + laneM) * 40 + hi * 8];
            b.h[0] = *(const ushort8*)(bp);
            b.h[1] = *(const ushort8*)(bp + 16);
            for (int s = 0; s < 2; ++s)
                acc[s][n] = wmma_bf16(a[s], b, acc[s][n]);
        }
        __syncthreads();
    }

    // ---- epilogue: bias + store ----
    for (int n = 0; n < 4; ++n) {
        int col  = n0 + wn * 64 + n * 16 + laneM;
        float bv = bias[col];
        for (int s = 0; s < 2; ++s)
            for (int r = 0; r < 8; ++r) {
                int row   = m0 + wm * 32 + s * 16 + r + hi * 8;
                float val = acc[s][n][r] + bv;
                if (OUT_F32) ((float*)outv)[(size_t)row * N + col] = val;
                else         ((u16*)outv)[(size_t)row * N + col]   = f2bf(val);
            }
    }
}

// =====================================================================
// Flash attention:  O = softmax(Q K^T / 8) V  per (batch, head).
// Q': [B*2048, 1024] bf16, head h occupies cols h*64..h*64+63. Same for K', V'.
// Block: 256 threads (8 waves) handles a 128-row Q chunk of one (b,h).
// Online softmax over KV in chunks of 64. K chunk staged via async copy.
// =====================================================================
__global__ __launch_bounds__(256)
void attn_wmma_kernel(const u16* __restrict__ Qp, const u16* __restrict__ Kp,
                      const u16* __restrict__ Vp, u16* __restrict__ Op) {
    __shared__ u16 Ks[64 * 72];       // K chunk [kv][d], stride 72 (144B, 16B aligned)
    __shared__ u16 Vt[64 * 72];       // V chunk transposed [d][kv]
    __shared__ u16 Ps[8 * 16 * 72];   // per-wave P slab for C-layout -> A-layout

    const int tid   = threadIdx.x;
    const int wave  = tid >> 5;
    const int lane  = tid & 31;
    const int laneM = lane & 15;
    const int hi    = lane >> 4;

    const int qc = blockIdx.x;   // q chunk   0..15
    const int h  = blockIdx.y;   // head      0..15
    const int b  = blockIdx.z;   // batch     0..7

    // ---- Q fragments for this wave's 16 rows (direct from global, bf16) ----
    const u16* qrow =
        Qp + ((size_t)(b * 2048 + qc * 128 + wave * 16 + laneM) * 1024 + h * 64);
    Frag qa[2];
    for (int j = 0; j < 2; ++j) {
        qa[j].h[0] = *(const ushort8*)(qrow + j * 32 + hi * 8);
        qa[j].h[1] = *(const ushort8*)(qrow + j * 32 + hi * 8 + 16);
    }

    v8f o[4];
    for (int n = 0; n < 4; ++n) o[n] = v8zero();
    float mrun[8], lrun[8];
    for (int r = 0; r < 8; ++r) { mrun[r] = -3.0e38f; lrun[r] = 0.f; }

    const float sc = 0.125f;   // 1/sqrt(d_k=64)

    for (int kv0 = 0; kv0 < 1024; kv0 += 64) {
        // ---- stage K chunk [64][64] row-major via async global->LDS copy ----
        for (int i = 0; i < 2; ++i) {
            int ch  = tid + i * 256;       // 512 chunks of 8 halves
            int row = ch >> 3;
            int off = (ch & 7) << 3;
            async_g2l_b128((unsigned)(uintptr_t)&Ks[row * 72 + off],
                           Kp + ((size_t)(b * 1024 + kv0 + row) * 1024 + h * 64 + off));
        }
        // ---- stage V chunk transposed -> Vt[d][kv] (needs VALU transpose) ----
        for (int i = 0; i < 2; ++i) {
            int ch  = tid + i * 256;
            int row = ch >> 3;             // kv
            int off = (ch & 7) << 3;       // d base
            ushort8 vv = *(const ushort8*)(
                Vp + ((size_t)(b * 1024 + kv0 + row) * 1024 + h * 64 + off));
            for (int e = 0; e < 8; ++e) Vt[(off + e) * 72 + row] = vv[e];
        }
        wait_async0();
        __syncthreads();

        // ---- S = Q @ K^T : 4 tiles of 16 kv columns ----
        v8f s[4];
        for (int n = 0; n < 4; ++n) {
            v8f c = v8zero();
            for (int j = 0; j < 2; ++j) {
                Frag bfr;
                const u16* bp = &Ks[(n * 16 + laneM) * 72 + j * 32 + hi * 8];
                bfr.h[0] = *(const ushort8*)(bp);
                bfr.h[1] = *(const ushort8*)(bp + 16);
                c = wmma_bf16(qa[j], bfr, c);
            }
            s[n] = c;
        }

        // ---- online softmax (rows split: lanes 0-15 -> r, lanes 16-31 -> r+8) ----
        float corr[8];
        for (int r = 0; r < 8; ++r) {
            float x = fmaxf(fmaxf(s[0][r], s[1][r]), fmaxf(s[2][r], s[3][r])) * sc;
            for (int m = 1; m < 16; m <<= 1) x = fmaxf(x, __shfl_xor(x, m, 32));
            float nm = fmaxf(mrun[r], x);
            corr[r]  = __expf(mrun[r] - nm);
            mrun[r]  = nm;
        }
        u16* pw = &Ps[wave * (16 * 72)];
        float rs[8];
        for (int r = 0; r < 8; ++r) rs[r] = 0.f;
        for (int n = 0; n < 4; ++n)
            for (int r = 0; r < 8; ++r) {
                float p = __expf(s[n][r] * sc - mrun[r]);
                rs[r] += p;
                pw[(r + hi * 8) * 72 + n * 16 + laneM] = f2bf(p);
            }
        for (int r = 0; r < 8; ++r) {
            float x = rs[r];
            for (int m = 1; m < 16; m <<= 1) x += __shfl_xor(x, m, 32);
            lrun[r] = lrun[r] * corr[r] + x;
            for (int n = 0; n < 4; ++n) o[n][r] *= corr[r];
        }

        // ---- O += P @ V ----  (P reloaded from wave-private LDS in A-layout)
        Frag pa[2];
        for (int j = 0; j < 2; ++j) {
            const u16* pp = &pw[laneM * 72 + j * 32 + hi * 8];
            pa[j].h[0] = *(const ushort8*)(pp);
            pa[j].h[1] = *(const ushort8*)(pp + 16);
        }
        for (int n = 0; n < 4; ++n)
            for (int j = 0; j < 2; ++j) {
                Frag bfr;
                const u16* bp = &Vt[(n * 16 + laneM) * 72 + j * 32 + hi * 8];
                bfr.h[0] = *(const ushort8*)(bp);
                bfr.h[1] = *(const ushort8*)(bp + 16);
                o[n] = wmma_bf16(pa[j], bfr, o[n]);
            }
        __syncthreads();
    }

    // ---- finalize: divide by l, store bf16 to AttnOut (concat layout) ----
    for (int n = 0; n < 4; ++n)
        for (int r = 0; r < 8; ++r) {
            int qrowG = qc * 128 + wave * 16 + r + hi * 8;
            float val = o[n][r] / lrun[r];
            Op[(size_t)(b * 2048 + qrowG) * 1024 + h * 64 + n * 16 + laneM] = f2bf(val);
        }
}

// =====================================================================
// Launcher
// =====================================================================
extern "C" void kernel_launch(void* const* d_in, const int* in_sizes, int n_in,
                              void* d_out, int out_size, void* d_ws, size_t ws_size,
                              hipStream_t stream) {
    const float* q  = (const float*)d_in[0];   // [8,2048,1024]
    const float* k  = (const float*)d_in[1];   // [8,1024,512]
    const float* v  = (const float*)d_in[2];   // [8,1024,512]
    const float* Wq = (const float*)d_in[3];
    const float* bq = (const float*)d_in[4];
    const float* Wk = (const float*)d_in[5];
    const float* bk = (const float*)d_in[6];
    const float* Wv = (const float*)d_in[7];
    const float* bv = (const float*)d_in[8];
    const float* Wo = (const float*)d_in[9];
    const float* bo = (const float*)d_in[10];

    // workspace layout (bf16 as u16), ~96 MB total
    u16* Qp = (u16*)d_ws;                  // [16384, 1024]
    u16* Kp = Qp + (size_t)16384 * 1024;   // [ 8192, 1024]
    u16* Vp = Kp + (size_t)8192  * 1024;   // [ 8192, 1024]
    u16* Ao = Vp + (size_t)8192  * 1024;   // [16384, 1024]

    dim3 blk(256);

    // projections (f32 in, bf16 out); N tiles of 128
    gemm_wmma_kernel<true, false><<<dim3(8, 128), blk, 0, stream>>>(
        (const void*)q, Wq, bq, (void*)Qp, 16384, 1024, 1024);
    gemm_wmma_kernel<true, false><<<dim3(8, 64), blk, 0, stream>>>(
        (const void*)k, Wk, bk, (void*)Kp, 8192, 1024, 512);
    gemm_wmma_kernel<true, false><<<dim3(8, 64), blk, 0, stream>>>(
        (const void*)v, Wv, bv, (void*)Vp, 8192, 1024, 512);

    // attention: grid = (q-chunks, heads, batch)
    attn_wmma_kernel<<<dim3(16, 16, 8), blk, 0, stream>>>(Qp, Kp, Vp, Ao);

    // output projection (bf16 in via async-LDS, f32 out -> d_out)
    gemm_wmma_kernel<false, true><<<dim3(8, 128), blk, 0, stream>>>(
        (const void*)Ao, Wo, bo, d_out, 16384, 1024, 1024);
}
